// EncoderBlock_52097953300796
// MI455X (gfx1250) — compile-verified, matrix operand order checked by reading, not run on hardware
//
#include <hip/hip_runtime.h>
#include <hip/hip_bf16.h>

// ---------------------------------------------------------------------------
// Types
// ---------------------------------------------------------------------------
typedef __bf16 bf16;
typedef __attribute__((ext_vector_type(16))) __bf16 bf16x16;
typedef __attribute__((ext_vector_type(8)))  float  f32x8;

#define B_  4
#define T_  2048
#define C_  1024
#define H_  16
#define HS_ 64
#define BT_ (B_ * T_)

// ---------------------------------------------------------------------------
// CDNA5 helpers
// ---------------------------------------------------------------------------
__device__ __forceinline__ f32x8 wmma_bf16(bf16x16 a, bf16x16 b, f32x8 c) {
    // emits v_wmma_f32_16x16x32_bf16
    return __builtin_amdgcn_wmma_f32_16x16x32_bf16(
        /*neg_a=*/false, a, /*neg_b=*/false, b,
        /*c_mod=*/(short)0, c, /*reuse_a=*/false, /*reuse_b=*/false);
}

// Async DMA: global memory -> LDS, 16 bytes per lane, tracked by ASYNCcnt.
// LDS byte address = low 32 bits of the generic shared pointer (ISA: LDS
// aperture truncates addr[31:0]).
__device__ __forceinline__ void async_copy_b128(void* lds, const void* gptr) {
    unsigned int       loff = (unsigned int)(unsigned long long)(uintptr_t)lds;
    unsigned long long ga   = (unsigned long long)(uintptr_t)gptr;
    asm volatile("global_load_async_to_lds_b128 %0, %1, off"
                 :: "v"(loff), "v"(ga) : "memory");
}
__device__ __forceinline__ void wait_async0() {
    asm volatile("s_wait_asynccnt 0x0" ::: "memory");
}

// Load one 16-value bf16 fragment for the 16x16x32 WMMA A/B layout.
// half 0 -> K {0..7, 16..23}, half 1 -> K {8..15, 24..31}; two ds_load_b128.
__device__ __forceinline__ bf16x16 load_frag(const bf16* base, int half) {
    bf16x16 f;
    *reinterpret_cast<uint4*>(&f)       = *reinterpret_cast<const uint4*>(base + half * 8);
    *(reinterpret_cast<uint4*>(&f) + 1) = *reinterpret_cast<const uint4*>(base + 16 + half * 8);
    return f;
}

// ---------------------------------------------------------------------------
// fp32 -> bf16 conversion (weights)
// ---------------------------------------------------------------------------
__global__ __launch_bounds__(256)
void cvt_bf16_kernel(const float* __restrict__ src, bf16* __restrict__ dst, int n) {
    int i = blockIdx.x * blockDim.x + threadIdx.x;
    int stride = gridDim.x * blockDim.x;
    for (; i < n; i += stride) dst[i] = (bf16)src[i];
}

// ---------------------------------------------------------------------------
// LayerNorm: one block per row, writes f32 and bf16 copies
// ---------------------------------------------------------------------------
__global__ __launch_bounds__(256)
void ln_kernel(const float* __restrict__ x, const float* __restrict__ g,
               const float* __restrict__ b, float* __restrict__ yf,
               bf16* __restrict__ yb) {
    const int row = blockIdx.x;
    const float* xr = x + (size_t)row * C_;
    __shared__ float red[256];
    const int tid = threadIdx.x;

    float s = 0.f;
    for (int c = tid; c < C_; c += 256) s += xr[c];
    red[tid] = s; __syncthreads();
    for (int off = 128; off > 0; off >>= 1) {
        if (tid < off) red[tid] += red[tid + off];
        __syncthreads();
    }
    const float mu = red[0] * (1.0f / C_);
    __syncthreads();

    float v = 0.f;
    for (int c = tid; c < C_; c += 256) { float d = xr[c] - mu; v += d * d; }
    red[tid] = v; __syncthreads();
    for (int off = 128; off > 0; off >>= 1) {
        if (tid < off) red[tid] += red[tid + off];
        __syncthreads();
    }
    const float rstd = rsqrtf(red[0] * (1.0f / C_) + 1e-5f);

    for (int c = tid; c < C_; c += 256) {
        float y = (xr[c] - mu) * rstd * g[c] + b[c];
        yf[(size_t)row * C_ + c] = y;
        yb[(size_t)row * C_ + c] = (bf16)y;
    }
}

// ---------------------------------------------------------------------------
// bf16 WMMA GEMM: C[M,N] = A[M,K] * B[K,N] (+bias) (ReLU) (+resid)
// Block tile 128x128, 8 waves (2x4), wave tile 64x32 = 4x2 WMMA accumulators.
// Double-buffered LDS; A tile staged via async DMA to LDS, B staged through
// registers (transpose to n-major so fragment loads are contiguous b128).
// ---------------------------------------------------------------------------
#define GBM 128
#define GBN 128
#define GBK 32
#define LPAD 8

template<int HAS_BIAS, int RELU, int HAS_RESID, int OUTF, int OUTB>
__global__ __launch_bounds__(256)
void gemm_bf16_kernel(const bf16* __restrict__ A, const bf16* __restrict__ Bm,
                      const float* __restrict__ bias, const float* __restrict__ resid,
                      float* __restrict__ Cf, bf16* __restrict__ Cb,
                      int M, int N, int K) {
    __shared__ bf16 As[2][GBM][GBK + LPAD];   // row-major A tiles
    __shared__ bf16 Bs[2][GBN][GBK + LPAD];   // transposed B tiles (n-major)

    const int ntn = N / GBN;
    const int m0 = (blockIdx.x / ntn) * GBM;
    const int n0 = (blockIdx.x % ntn) * GBN;

    const int tid  = threadIdx.x;
    const int wid  = tid >> 5;      // 0..7
    const int lane = tid & 31;
    const int half = lane >> 4;
    const int lr   = lane & 15;
    const int wm   = wid >> 2;      // 0..1 -> 64 rows each
    const int wn   = wid & 3;       // 0..3 -> 32 cols each

    f32x8 acc[4][2] = {};

    auto stage = [&](int buf, int k0) {
        // A tile: 128x32, async DMA, 2 x b128 per thread
#pragma unroll
        for (int i = 0; i < 2; ++i) {
            int c  = tid + i * 256;            // 512 chunks of 8 bf16
            int ar = c >> 2;
            int ac = (c & 3) * 8;
            async_copy_b128(&As[buf][ar][ac],
                            A + (size_t)(m0 + ar) * K + k0 + ac);
        }
        // B tile: read 8 along N (coalesced), scatter transposed to Bs[n][k]
#pragma unroll
        for (int i = 0; i < 2; ++i) {
            int c  = tid + i * 256;
            int kr = c >> 4;                   // 0..31
            int nc = (c & 15) * 8;
            bf16 tmp[8];
            *reinterpret_cast<uint4*>(tmp) =
                *reinterpret_cast<const uint4*>(Bm + (size_t)(k0 + kr) * N + n0 + nc);
#pragma unroll
            for (int j = 0; j < 8; ++j) Bs[buf][nc + j][kr] = tmp[j];
        }
    };

    stage(0, 0);
    wait_async0();
    __syncthreads();

    const int nk = K / GBK;
    for (int t = 0; t < nk; ++t) {
        const int buf = t & 1;
        if (t + 1 < nk) stage(buf ^ 1, (t + 1) * GBK);   // overlap with WMMA

        bf16x16 afrag[4], bfrag[2];
#pragma unroll
        for (int i = 0; i < 4; ++i)
            afrag[i] = load_frag(&As[buf][wm * 64 + i * 16 + lr][0], half);
#pragma unroll
        for (int j = 0; j < 2; ++j)
            bfrag[j] = load_frag(&Bs[buf][wn * 32 + j * 16 + lr][0], half);
#pragma unroll
        for (int i = 0; i < 4; ++i)
#pragma unroll
            for (int j = 0; j < 2; ++j)
                acc[i][j] = wmma_bf16(afrag[i], bfrag[j], acc[i][j]);

        wait_async0();          // our async copies into buf^1 have landed
        __syncthreads();        // everyone's staging visible; prev reads done
    }

    // ---- epilogue (flags are template constants: no runtime branches)
#pragma unroll
    for (int i = 0; i < 4; ++i) {
#pragma unroll
        for (int j = 0; j < 2; ++j) {
            const int colg = n0 + wn * 32 + j * 16 + lr;
            const float bia = HAS_BIAS ? bias[colg] : 0.0f;
#pragma unroll
            for (int rr = 0; rr < 8; ++rr) {
                const int rowg = m0 + wm * 64 + i * 16 + half * 8 + rr;
                float v = acc[i][j][rr] + bia;
                if (RELU) v = fmaxf(v, 0.0f);
                if (HAS_RESID) v += resid[(size_t)rowg * N + colg];
                if (OUTF) Cf[(size_t)rowg * N + colg] = v;
                if (OUTB) Cb[(size_t)rowg * N + colg] = (bf16)v;
            }
        }
    }
}

// ---------------------------------------------------------------------------
// Flash attention per (b, h): S = QK^T/sqrt(C), online softmax, O = P V.
// Block = 128 threads (4 waves); block handles 64 query rows, wave owns 16.
// Q/K tiles staged via async DMA (row-major both sides); V transposed via
// registers. Q/K/V stored bf16 in [B*T, H*HS] layout (col = h*HS + d).
// ---------------------------------------------------------------------------
__global__ __launch_bounds__(128)
void attn_kernel(const bf16* __restrict__ Q, const bf16* __restrict__ Kp,
                 const bf16* __restrict__ V, bf16* __restrict__ O) {
    const int bh = blockIdx.x;
    const int b  = bh / H_;
    const int h  = bh % H_;
    const int q0 = blockIdx.y * 64;

    const size_t rowBase = (size_t)b * T_;
    const int    colBase = h * HS_;

    __shared__ bf16  Qs [64][HS_ + LPAD];
    __shared__ bf16  Ks [64][HS_ + LPAD];   // row-major: serves as B^T for QK^T
    __shared__ bf16  Vts[HS_][64 + LPAD];   // transposed: B^T for P*V
    __shared__ float Ss [4][16][64];
    __shared__ bf16  Ps [4][16][64];
    __shared__ float Fs [4][16];

    const int tid  = threadIdx.x;
    const int wid  = tid >> 5;
    const int lane = tid & 31;
    const int half = lane >> 4;
    const int lr   = lane & 15;

    // ---- Q tile (64x64): async DMA, 4 x b128 per thread
#pragma unroll
    for (int i = 0; i < 4; ++i) {
        int c = tid + i * 128;              // 512 chunks of 8
        int r = c >> 3;
        int cc = (c & 7) * 8;
        async_copy_b128(&Qs[r][cc],
                        Q + (rowBase + q0 + r) * C_ + colBase + cc);
    }

    f32x8 o[4] = {};
    float m_run = -1e30f;
    float l_run = 0.0f;
    const float scale = 0.03125f;           // 1/sqrt(C) = 1/32

    for (int s0 = 0; s0 < T_; s0 += 64) {
        __syncthreads();                    // previous tile fully consumed
        // ---- stage K tile (async, row-major) and V tile (transposed)
#pragma unroll
        for (int i = 0; i < 4; ++i) {
            int c = tid + i * 128;
            int r = c >> 3;
            int cc = (c & 7) * 8;
            async_copy_b128(&Ks[r][cc],
                            Kp + (rowBase + s0 + r) * C_ + colBase + cc);
            bf16 tmp[8];
            *reinterpret_cast<uint4*>(tmp) =
                *reinterpret_cast<const uint4*>(V + (rowBase + s0 + r) * C_ + colBase + cc);
#pragma unroll
            for (int j = 0; j < 8; ++j) Vts[cc + j][r] = tmp[j];
        }
        wait_async0();                      // K (and first-iter Q) landed
        __syncthreads();

        // ---- S tile = Q(wave's 16 rows) x K^T (64 cols), K-dim = 64
        f32x8 sacc[4] = {};
#pragma unroll
        for (int kk = 0; kk < HS_; kk += 32) {
            bf16x16 aq = load_frag(&Qs[wid * 16 + lr][kk], half);
#pragma unroll
            for (int j = 0; j < 4; ++j) {
                bf16x16 bk = load_frag(&Ks[j * 16 + lr][kk], half);
                sacc[j] = wmma_bf16(aq, bk, sacc[j]);
            }
        }
        // scatter S to per-wave scratch with scale
#pragma unroll
        for (int j = 0; j < 4; ++j)
#pragma unroll
            for (int rr = 0; rr < 8; ++rr)
                Ss[wid][half * 8 + rr][j * 16 + lr] = sacc[j][rr] * scale;

        // ---- online softmax: lane handles row lr (halves duplicate)
        const int row = lr;
        float tmax = -1e30f;
        for (int c2 = 0; c2 < 64; ++c2) tmax = fmaxf(tmax, Ss[wid][row][c2]);
        const float m_new = fmaxf(m_run, tmax);
        const float corr  = __expf(m_run - m_new);
        float lsum = 0.0f;
        for (int c2 = 0; c2 < 64; ++c2) {
            float p = __expf(Ss[wid][row][c2] - m_new);
            lsum += p;
            Ps[wid][row][c2] = (bf16)p;
        }
        l_run = l_run * corr + lsum;
        m_run = m_new;
        Fs[wid][row] = corr;

        // ---- rescale existing O accumulators by per-row corr
#pragma unroll
        for (int rr = 0; rr < 8; ++rr) {
            const float f = Fs[wid][half * 8 + rr];
#pragma unroll
            for (int j = 0; j < 4; ++j) o[j][rr] *= f;
        }

        // ---- O += P x V   (K-dim = 64 keys)
#pragma unroll
        for (int kk = 0; kk < 64; kk += 32) {
            bf16x16 ap = load_frag(&Ps[wid][lr][kk], half);
#pragma unroll
            for (int j = 0; j < 4; ++j) {
                bf16x16 bv = load_frag(&Vts[j * 16 + lr][kk], half);
                o[j] = wmma_bf16(ap, bv, o[j]);
            }
        }
    }

    // ---- normalize by running denominator, store bf16
    Fs[wid][lr] = l_run;
#pragma unroll
    for (int rr = 0; rr < 8; ++rr) {
        const float inv = 1.0f / Fs[wid][half * 8 + rr];
        const int rowg = q0 + wid * 16 + half * 8 + rr;
#pragma unroll
        for (int j = 0; j < 4; ++j) {
            const int colg = colBase + j * 16 + lr;
            O[(rowBase + rowg) * C_ + colg] = (bf16)(o[j][rr] * inv);
        }
    }
}

// ---------------------------------------------------------------------------
// Host-side orchestration
// ---------------------------------------------------------------------------
extern "C" void kernel_launch(void* const* d_in, const int* in_sizes, int n_in,
                              void* d_out, int out_size, void* d_ws, size_t ws_size,
                              hipStream_t stream) {
    const float* x      = (const float*)d_in[0];
    const float* wq     = (const float*)d_in[1];
    const float* wk     = (const float*)d_in[2];
    const float* wv     = (const float*)d_in[3];
    const float* w_proj = (const float*)d_in[4];
    const float* b_proj = (const float*)d_in[5];
    const float* ln1_g  = (const float*)d_in[6];
    const float* ln1_b  = (const float*)d_in[7];
    const float* ln2_g  = (const float*)d_in[8];
    const float* ln2_b  = (const float*)d_in[9];
    const float* w1     = (const float*)d_in[10];
    const float* b1     = (const float*)d_in[11];
    const float* w2     = (const float*)d_in[12];
    const float* b2     = (const float*)d_in[13];
    float* out = (float*)d_out;

    // ---- carve workspace
    char* p = (char*)d_ws;
    auto carve = [&](size_t bytes) -> void* {
        void* r = (void*)p;
        p += (bytes + 255) & ~(size_t)255;
        return r;
    };
    const size_t BTC = (size_t)BT_ * C_;
    float* x1f = (float*)carve(BTC * 4);
    float* x2f = (float*)carve(BTC * 4);
    float* x3f = (float*)carve(BTC * 4);
    bf16*  x1b = (bf16*)carve(BTC * 2);
    bf16*  x3b = (bf16*)carve(BTC * 2);
    bf16*  Qb  = (bf16*)carve(BTC * 2);
    bf16*  Kb  = (bf16*)carve(BTC * 2);
    bf16*  Vb  = (bf16*)carve(BTC * 2);
    bf16*  Ab  = (bf16*)carve(BTC * 2);               // attention output
    bf16*  h1b = (bf16*)carve(BTC * 4 * 2);           // ReLU(x3 w1 + b1)
    bf16*  wqb = (bf16*)carve((size_t)C_ * C_ * 2);
    bf16*  wkb = (bf16*)carve((size_t)C_ * C_ * 2);
    bf16*  wvb = (bf16*)carve((size_t)C_ * C_ * 2);
    bf16*  wpb = (bf16*)carve((size_t)C_ * C_ * 2);
    bf16*  w1b = (bf16*)carve((size_t)C_ * 4 * C_ * 2);
    bf16*  w2b = (bf16*)carve((size_t)C_ * 4 * C_ * 2);
    (void)ws_size; (void)n_in; (void)in_sizes; (void)out_size;

    // ---- weight conversion fp32 -> bf16
    auto cvt = [&](const float* s, bf16* d, int n) {
        cvt_bf16_kernel<<<(n + 255) / 256, 256, 0, stream>>>(s, d, n);
    };
    cvt(wq,     wqb, C_ * C_);
    cvt(wk,     wkb, C_ * C_);
    cvt(wv,     wvb, C_ * C_);
    cvt(w_proj, wpb, C_ * C_);
    cvt(w1,     w1b, C_ * 4 * C_);
    cvt(w2,     w2b, C_ * 4 * C_);

    // ---- LN1
    ln_kernel<<<BT_, 256, 0, stream>>>(x, ln1_g, ln1_b, x1f, x1b);

    // ---- Q/K/V projections (wq[h,c,d] flattens to [C x (h*HS+d)] columns)
    const int gQKV = (BT_ / GBM) * (C_ / GBN);
    gemm_bf16_kernel<0,0,0,0,1><<<gQKV, 256, 0, stream>>>(
        x1b, wqb, nullptr, nullptr, nullptr, Qb, BT_, C_, C_);
    gemm_bf16_kernel<0,0,0,0,1><<<gQKV, 256, 0, stream>>>(
        x1b, wkb, nullptr, nullptr, nullptr, Kb, BT_, C_, C_);
    gemm_bf16_kernel<0,0,0,0,1><<<gQKV, 256, 0, stream>>>(
        x1b, wvb, nullptr, nullptr, nullptr, Vb, BT_, C_, C_);

    // ---- attention
    {
        dim3 grid(B_ * H_, T_ / 64);
        attn_kernel<<<grid, 128, 0, stream>>>(Qb, Kb, Vb, Ab);
    }

    // ---- output projection + bias + residual(x1) -> x2 (f32)
    gemm_bf16_kernel<1,0,1,1,0><<<gQKV, 256, 0, stream>>>(
        Ab, wpb, b_proj, x1f, x2f, nullptr, BT_, C_, C_);

    // ---- LN2 -> x3 (f32 + bf16)
    ln_kernel<<<BT_, 256, 0, stream>>>(x2f, ln2_g, ln2_b, x3f, x3b);

    // ---- FFN
    const int gFF1 = (BT_ / GBM) * (4 * C_ / GBN);
    gemm_bf16_kernel<1,1,0,0,1><<<gFF1, 256, 0, stream>>>(
        x3b, w1b, b1, nullptr, nullptr, h1b, BT_, 4 * C_, C_);
    gemm_bf16_kernel<1,0,1,1,0><<<gQKV, 256, 0, stream>>>(
        h1b, w2b, b2, x3f, out, nullptr, BT_, C_, 4 * C_);
}
